// attblock_28810640622053
// MI455X (gfx1250) — matrix operand order repaired and checked by reading, not compiled, not run on hardware
//
#include <hip/hip_runtime.h>
#include <hip/hip_bf16.h>
#include <math.h>

typedef __attribute__((ext_vector_type(16))) __bf16      v16bf;
typedef __attribute__((ext_vector_type(8)))  float       v8f;
typedef __attribute__((ext_vector_type(8)))  unsigned int v8u;

// Problem constants (reference: B=8, H=W=64, C=256, ws=8, nh=8)
#define C_DIM   256
#define N_HEADS 8
#define HD      32
#define N_WIN   64          // tokens per window (8*8)
#define BW      512         // total windows = 8 * 8 * 8
#define MROWS   32768       // BW * N_WIN

// ---- CDNA5 async global->LDS copy (guarded; fallback = manual staging) ----
#if defined(__has_builtin)
#  if __has_builtin(__builtin_amdgcn_global_load_async_to_lds_b128) && \
      __has_builtin(__builtin_amdgcn_s_wait_asynccnt)
#    define USE_ASYNC_COPY 1
#  endif
#endif
#ifndef USE_ASYNC_COPY
#  define USE_ASYNC_COPY 0
#endif

#define AS1 __attribute__((address_space(1)))
#define AS3 __attribute__((address_space(3)))

// pointee type required by the async-to-LDS builtins (per clang diagnostic):
// __attribute__((vector_size(16))) int
typedef int b128_t __attribute__((vector_size(16)));

// ---------- helpers ----------

__device__ __forceinline__ unsigned short f2bf(float f) {
  unsigned u = __builtin_bit_cast(unsigned, f);
  u += 0x7FFFu + ((u >> 16) & 1u);        // round-to-nearest-even
  return (unsigned short)(u >> 16);
}

// window-partitioned row -> row in [B, H*W] token space
__device__ __forceinline__ int win_row_to_global(int r) {
  int bw = r >> 6, n = r & 63;
  int b = bw >> 6, wh = (bw >> 3) & 7, ww = bw & 7;
  int ii = n >> 3, jj = n & 7;
  return b * 4096 + (wh * 8 + ii) * 64 + (ww * 8 + jj);
}

// 16-bit A-matrix 16x32 fragment (CDNA5 ISA 7.12.2): memory layout [m][k], stride in elems
__device__ __forceinline__ v16bf load_a_frag(const unsigned short* p, int m0, int stride) {
  int lane = threadIdx.x & 31;
  int r = m0 + (lane & 15);
  int kofs = (lane < 16) ? 0 : 8;
  v8u u;
#pragma unroll
  for (int j = 0; j < 8; ++j) {
    int k = (j < 4) ? (kofs + 2 * j) : (16 + kofs + 2 * (j - 4));
    u[j] = *(const unsigned int*)(p + r * stride + k);   // bf16 pair, 4B aligned
  }
  return __builtin_bit_cast(v16bf, u);
}

// 16-bit B-matrix 32x16 fragment: memory layout [n][k] (B element (k,n)), stride in elems
__device__ __forceinline__ v16bf load_b_frag(const unsigned short* p, int n0, int stride) {
  int lane = threadIdx.x & 31;
  int r = n0 + (lane & 15);
  int kofs = (lane < 16) ? 0 : 8;
  v8u u;
#pragma unroll
  for (int j = 0; j < 8; ++j) {
    int k = (j < 4) ? (kofs + 2 * j) : (16 + kofs + 2 * (j - 4));
    u[j] = *(const unsigned int*)(p + r * stride + k);
  }
  return __builtin_bit_cast(v16bf, u);
}

#define WMMA_BF16(A, B, Cacc) \
  __builtin_amdgcn_wmma_f32_16x16x32_bf16(false, (A), false, (B), (short)0, (Cacc), false, false)

// ---------- kernel 1: convert all 8 weight matrices f32 -> bf16 ----------

struct WPtrs { const float* w[8]; };

__global__ void convert_weights_kernel(WPtrs wp, unsigned short* __restrict__ dst) {
  int idx = blockIdx.x * 256 + threadIdx.x;        // 8 * 65536 total
  int m = idx >> 16, o = idx & 0xFFFF;
  dst[idx] = f2bf(wp.w[m][o]);
}

// ---------- kernel 2: projection GEMM  dst[bw,head,n,hd] = (win(x) @ W + b) * scale ----------
// grid (512, 2), block 128 (4 waves). Tile 64(M) x 128(N), K-step 32.

__global__ __launch_bounds__(128) void proj_qkv_kernel(
    const float* __restrict__ x, const unsigned short* __restrict__ w,
    const float* __restrict__ bias, unsigned short* __restrict__ dst, float scale)
{
  __shared__ __align__(16) unsigned short lA[64 * 34];
  __shared__ __align__(16) unsigned short lB[128 * 34];
  int m0 = blockIdx.x * 64;
  int n0 = blockIdx.y * 128;
  int tid = threadIdx.x, wave = tid >> 5, lane = tid & 31;

  v8f acc[8];
#pragma unroll
  for (int t = 0; t < 8; ++t) acc[t] = {};

  for (int k0 = 0; k0 < 256; k0 += 32) {
    // A tile 64x32 (f32 global -> bf16 LDS), window row remap on the load side
#pragma unroll
    for (int it = 0; it < 16; ++it) {
      int e = tid + it * 128;
      int r = e >> 5, c = e & 31;
      int g = win_row_to_global(m0 + r);
      lA[r * 34 + c] = f2bf(x[(size_t)g * 256 + k0 + c]);
    }
    // B tile 32x128, transposed into [n][k]
#pragma unroll
    for (int it = 0; it < 32; ++it) {
      int e = tid + it * 128;
      int n = e & 127, k = e >> 7;
      lB[n * 34 + k] = w[(k0 + k) * 256 + n0 + n];
    }
    __syncthreads();
    v16bf af = load_a_frag(lA, wave * 16, 34);
#pragma unroll
    for (int t = 0; t < 8; ++t) {
      v16bf bf = load_b_frag(lB, t * 16, 34);
      acc[t] = WMMA_BF16(af, bf, acc[t]);
    }
    __syncthreads();
  }

  int rlo = m0 + wave * 16 + ((lane < 16) ? 0 : 8);
#pragma unroll
  for (int t = 0; t < 8; ++t) {
    int col = n0 + t * 16 + (lane & 15);
    float bb = bias[col];
    int head = col >> 5, hd = col & 31;
#pragma unroll
    for (int i = 0; i < 8; ++i) {
      int row = rlo + i;
      int bw = row >> 6, n = row & 63;
      float v = (acc[t][i] + bb) * scale;
      dst[(size_t)((bw * 8 + head) * 64 + n) * 32 + hd] = f2bf(v);
    }
  }
}

// ---------- kernel 3: attention per (window, head), all 3 query sets ----------
// block = 128 threads (4 waves, each owns a 16-row query stripe)
// K tile staged via async global->LDS b128 (stride 40 keeps rows 16B-aligned)

__global__ __launch_bounds__(128) void attn_kernel(
    const unsigned short* __restrict__ q1, const unsigned short* __restrict__ q2,
    const unsigned short* __restrict__ q3, const unsigned short* __restrict__ kbuf,
    const unsigned short* __restrict__ vbuf, const float* __restrict__ rpb,
    unsigned short* __restrict__ atto)
{
  __shared__ __align__(16) unsigned short lK[64 * 40];   // [key][hd], 80B row pitch
  __shared__ __align__(16) unsigned short lV[32 * 66];   // [hd][key] (transposed)
  __shared__ __align__(16) unsigned short lP[64 * 66];   // [q][key] bf16 probs
  __shared__ float lBias[225];

  int blk = blockIdx.x;            // bw*8 + head
  int head = blk & 7;
  int bw = blk >> 3;
  int tid = threadIdx.x, wave = tid >> 5, lane = tid & 31;

  const unsigned short* kg = kbuf + (size_t)blk * 2048;
  const unsigned short* vg = vbuf + (size_t)blk * 2048;

#if USE_ASYNC_COPY
  // 64x32 bf16 K tile = 256 x 16B chunks; chunk -> (key, quarter-row)
#pragma unroll
  for (int it = 0; it < 2; ++it) {
    int e = tid + it * 128;
    int key = e >> 2, part = e & 3;
    __builtin_amdgcn_global_load_async_to_lds_b128(
        (AS1 b128_t*)(kg + key * 32 + part * 8),
        (AS3 b128_t*)(&lK[key * 40 + part * 8]), 0, 0);
  }
#else
#pragma unroll
  for (int it = 0; it < 16; ++it) {
    int e = tid + it * 128;
    int r = e >> 5, c = e & 31;
    lK[r * 40 + c] = kg[e];
  }
#endif
  // V transpose must go through VGPRs anyway
#pragma unroll
  for (int it = 0; it < 16; ++it) {
    int e = tid + it * 128;
    int r = e >> 5, c = e & 31;
    lV[c * 66 + r] = vg[e];
  }
  for (int e = tid; e < 225; e += 128) lBias[e] = rpb[e * 8 + head];
#if USE_ASYNC_COPY
  __builtin_amdgcn_s_wait_asynccnt(0);
#endif
  __syncthreads();

  const unsigned short* qs[3] = {q1, q2, q3};
  for (int qsel = 0; qsel < 3; ++qsel) {
    const unsigned short* qg = qs[qsel] + (size_t)blk * 2048;
    v16bf qf = load_a_frag(qg, wave * 16, 32);       // q fragments straight from global

    v8f s[4];
#pragma unroll
    for (int t = 0; t < 4; ++t) {
      v16bf kf = load_b_frag(lK, t * 16, 40);
      v8f z = {};
      s[t] = WMMA_BF16(qf, kf, z);                   // S = q @ k^T (q pre-scaled)
    }

    // row softmax, then ADD bias AFTER normalization (per reference), emit bf16 P
#pragma unroll
    for (int i = 0; i < 8; ++i) {
      float a[4];
#pragma unroll
      for (int t = 0; t < 4; ++t) a[t] = s[t][i];
      float m = fmaxf(fmaxf(a[0], a[1]), fmaxf(a[2], a[3]));
#pragma unroll
      for (int off = 1; off < 16; off <<= 1) m = fmaxf(m, __shfl_xor(m, off, 16));
      float e_[4], sum = 0.f;
#pragma unroll
      for (int t = 0; t < 4; ++t) { e_[t] = __expf(a[t] - m); sum += e_[t]; }
#pragma unroll
      for (int off = 1; off < 16; off <<= 1) sum += __shfl_xor(sum, off, 16);
      float inv = 1.0f / sum;
      int rq = wave * 16 + i + ((lane < 16) ? 0 : 8);
      int i1 = rq >> 3, j1 = rq & 7;
#pragma unroll
      for (int t = 0; t < 4; ++t) {
        int ck = t * 16 + (lane & 15);
        int i2 = ck >> 3, j2 = ck & 7;
        float p = e_[t] * inv + lBias[(i1 - i2 + 7) * 15 + (j1 - j2 + 7)];
        lP[rq * 66 + ck] = f2bf(p);
      }
    }

    // O = P @ V : M=64, K=64 (2 steps), N=32 (2 tiles) -- wave reads its own stripe
    v8f o[2]; o[0] = {}; o[1] = {};
#pragma unroll
    for (int ks = 0; ks < 2; ++ks) {
      v16bf pf = load_a_frag(lP + ks * 32, wave * 16, 66);
#pragma unroll
      for (int t = 0; t < 2; ++t) {
        v16bf vf = load_b_frag(lV + ks * 32, t * 16, 66);
        o[t] = WMMA_BF16(pf, vf, o[t]);
      }
    }

    unsigned short* op = atto + (size_t)qsel * ((size_t)MROWS * 256);
    int rlo = wave * 16 + ((lane < 16) ? 0 : 8);
#pragma unroll
    for (int t = 0; t < 2; ++t) {
      int col = head * 32 + t * 16 + (lane & 15);
#pragma unroll
      for (int i = 0; i < 8; ++i) {
        int row = bw * 64 + rlo + i;
        op[(size_t)row * 256 + col] = f2bf(o[t][i]);
      }
    }
    __syncthreads();
  }
}

// ---------- kernel 4: output projection, window_reverse folded into the store ----------
// A tile (bf16) staged via async global->LDS b128 when available

__global__ __launch_bounds__(128) void outproj_kernel(
    const unsigned short* __restrict__ a, const unsigned short* __restrict__ w,
    const float* __restrict__ bias, float* __restrict__ out)
{
  __shared__ __align__(16) unsigned short lA[64 * 40];   // [m][k], 80B row pitch
  __shared__ __align__(16) unsigned short lB[128 * 34];
  int m0 = blockIdx.x * 64, n0 = blockIdx.y * 128;
  int tid = threadIdx.x, wave = tid >> 5, lane = tid & 31;

  v8f acc[8];
#pragma unroll
  for (int t = 0; t < 8; ++t) acc[t] = {};

  for (int k0 = 0; k0 < 256; k0 += 32) {
#if USE_ASYNC_COPY
#pragma unroll
    for (int it = 0; it < 2; ++it) {
      int e = tid + it * 128;
      int r = e >> 2, part = e & 3;
      __builtin_amdgcn_global_load_async_to_lds_b128(
          (AS1 b128_t*)(a + (size_t)(m0 + r) * 256 + k0 + part * 8),
          (AS3 b128_t*)(&lA[r * 40 + part * 8]), 0, 0);
    }
#else
#pragma unroll
    for (int it = 0; it < 16; ++it) {
      int e = tid + it * 128;
      int r = e >> 5, c = e & 31;
      lA[r * 40 + c] = a[(size_t)(m0 + r) * 256 + k0 + c];
    }
#endif
#pragma unroll
    for (int it = 0; it < 32; ++it) {
      int e = tid + it * 128;
      int n = e & 127, k = e >> 7;
      lB[n * 34 + k] = w[(k0 + k) * 256 + n0 + n];
    }
#if USE_ASYNC_COPY
    __builtin_amdgcn_s_wait_asynccnt(0);
#endif
    __syncthreads();
    v16bf af = load_a_frag(lA, wave * 16, 40);
#pragma unroll
    for (int t = 0; t < 8; ++t) {
      v16bf bf = load_b_frag(lB, t * 16, 34);
      acc[t] = WMMA_BF16(af, bf, acc[t]);
    }
    __syncthreads();
  }

  int rlo = m0 + wave * 16 + ((lane < 16) ? 0 : 8);
#pragma unroll
  for (int t = 0; t < 8; ++t) {
    int col = n0 + t * 16 + (lane & 15);
    float bb = bias[col];
#pragma unroll
    for (int i = 0; i < 8; ++i) {
      int g = win_row_to_global(rlo + i);   // window_reverse
      out[(size_t)g * 256 + col] = acc[t][i] + bb;
    }
  }
}

// ---------- host launcher ----------

extern "C" void kernel_launch(void* const* d_in, const int* in_sizes, int n_in,
                              void* d_out, int out_size, void* d_ws, size_t ws_size,
                              hipStream_t stream) {
  const float* x1  = (const float*)d_in[0];
  const float* x2  = (const float*)d_in[1];
  const float* x3  = (const float*)d_in[2];
  const float* x4  = (const float*)d_in[3];
  const float* wq1 = (const float*)d_in[4];  const float* bq1 = (const float*)d_in[5];
  const float* wq2 = (const float*)d_in[6];  const float* bq2 = (const float*)d_in[7];
  const float* wq3 = (const float*)d_in[8];  const float* bq3 = (const float*)d_in[9];
  const float* wk  = (const float*)d_in[10]; const float* bk  = (const float*)d_in[11];
  const float* wv  = (const float*)d_in[12]; const float* bv  = (const float*)d_in[13];
  const float* wp1 = (const float*)d_in[14]; const float* bp1 = (const float*)d_in[15];
  const float* wp2 = (const float*)d_in[16]; const float* bp2 = (const float*)d_in[17];
  const float* wp3 = (const float*)d_in[18]; const float* bp3 = (const float*)d_in[19];
  const float* rpb = (const float*)d_in[20];
  float* out = (float*)d_out;

  // workspace layout: 8 bf16 weight mats (1MB) | q1,q2,q3,k,v (5x16MB) | atto (3x16MB)
  char* ws = (char*)d_ws;
  unsigned short* wbf = (unsigned short*)ws;
  unsigned short* qkv = (unsigned short*)(ws + (size_t)8 * 65536 * 2);
  const size_t plane = (size_t)MROWS * 256;
  unsigned short* q1b = qkv;
  unsigned short* q2b = qkv + plane;
  unsigned short* q3b = qkv + 2 * plane;
  unsigned short* kb  = qkv + 3 * plane;
  unsigned short* vb  = qkv + 4 * plane;
  unsigned short* atto = qkv + 5 * plane;

  WPtrs wp;
  wp.w[0] = wq1; wp.w[1] = wq2; wp.w[2] = wq3; wp.w[3] = wk;
  wp.w[4] = wv;  wp.w[5] = wp1; wp.w[6] = wp2; wp.w[7] = wp3;
  convert_weights_kernel<<<2048, 256, 0, stream>>>(wp, wbf);

  const float scale = 0.1767766952966369f;  // 32^-0.5, folded into q
  dim3 gg(512, 2);
  proj_qkv_kernel<<<gg, 128, 0, stream>>>(x2, wbf + 0 * 65536, bq1, q1b, scale);
  proj_qkv_kernel<<<gg, 128, 0, stream>>>(x3, wbf + 1 * 65536, bq2, q2b, scale);
  proj_qkv_kernel<<<gg, 128, 0, stream>>>(x4, wbf + 2 * 65536, bq3, q3b, scale);
  proj_qkv_kernel<<<gg, 128, 0, stream>>>(x1, wbf + 3 * 65536, bk,  kb,  1.0f);
  proj_qkv_kernel<<<gg, 128, 0, stream>>>(x1, wbf + 4 * 65536, bv,  vb,  1.0f);

  attn_kernel<<<BW * N_HEADS, 128, 0, stream>>>(q1b, q2b, q3b, kb, vb, rpb, atto);

  outproj_kernel<<<gg, 128, 0, stream>>>(atto,             wbf + 5 * 65536, bp1, out);
  outproj_kernel<<<gg, 128, 0, stream>>>(atto + plane,     wbf + 6 * 65536, bp2, out + plane);
  outproj_kernel<<<gg, 128, 0, stream>>>(atto + 2 * plane, wbf + 7 * 65536, bp3, out + 2 * plane);
}
